// ContrastiveLoss_29265907155300
// MI455X (gfx1250) — matrix operand order repaired and checked by reading, not compiled, not run on hardware
//
#include <hip/hip_runtime.h>
#include <hip/hip_bf16.h>

typedef __attribute__((ext_vector_type(16))) _Float16 v16h;
typedef __attribute__((ext_vector_type(8)))  _Float16 v8h;
typedef __attribute__((ext_vector_type(4)))  _Float16 v4h;
typedef __attribute__((ext_vector_type(8)))  float    v8f;

union V16 { v16h v; v8h h[2]; };

#define NROWS 8192
#define DIM   128
#define TILE  128
#define NT    (NROWS / TILE)   // 64 tiles per dimension

// ---------------------------------------------------------------------------
// Kernel 0: zero the per-row accumulators + label histogram (ws is poisoned
// with 0xAA once and never re-poisoned, and atomics accumulate, so this must
// run every call for determinism).
// ---------------------------------------------------------------------------
__global__ __launch_bounds__(256) void k_zero(float* __restrict__ rowSumExp,
                                              float* __restrict__ rowSumPos,
                                              int*   __restrict__ cnt) {
    int i = blockIdx.x * 256 + threadIdx.x;
    if (i < NROWS) {
        rowSumExp[i] = 0.0f;
        rowSumPos[i] = 0.0f;
    }
    if (i < 256) cnt[i] = 0;
}

// ---------------------------------------------------------------------------
// Kernel 1: row-normalize features (f32 -> f16), clamp norm at 1e-8 like the
// reference, and build the label histogram (n_pos(i) == cnt[label(i)]).
// One wave (32 lanes) per row: 4 floats per lane, shfl_xor tree reduction.
// ---------------------------------------------------------------------------
__global__ __launch_bounds__(256) void k_prep(const float* __restrict__ feat,
                                              const int*   __restrict__ labels,
                                              _Float16*    __restrict__ fn,
                                              int*         __restrict__ cnt) {
    const int row  = blockIdx.x * 8 + (threadIdx.x >> 5);
    const int lane = threadIdx.x & 31;

    const float4 v = ((const float4*)(feat + row * DIM))[lane];
    float ss = v.x * v.x + v.y * v.y + v.z * v.z + v.w * v.w;
    #pragma unroll
    for (int off = 16; off > 0; off >>= 1) ss += __shfl_xor(ss, off, 32);

    const float inv = 1.0f / fmaxf(sqrtf(ss), 1e-8f);

    v4h o;
    o[0] = (_Float16)(v.x * inv);
    o[1] = (_Float16)(v.y * inv);
    o[2] = (_Float16)(v.z * inv);
    o[3] = (_Float16)(v.w * inv);
    ((v4h*)(fn + row * DIM))[lane] = o;

    if (lane == 0) atomicAdd(&cnt[labels[row]], 1);
}

// ---------------------------------------------------------------------------
// Kernel 2: fused WMMA GEMM + epilogue.
// Block = 256 threads (8 waves) computes a 128x128 tile of sim = fn @ fn^T.
// Wave w owns rows [w*16, w*16+16) of the tile x all 128 columns
// (8 wave-tiles of 16x16, K looped in 4 chunks of 32).
// Epilogue (per reference, with T from d_in): logit = sim/T,
//   rowSumExp[i] += sum_j (label_i!=label_j) * exp(logit_ij)
//   rowSumPos[i] += sum_j (label_i==label_j) * logit_ij
// ---------------------------------------------------------------------------
__global__ __launch_bounds__(256) void k_gemm_fused(const _Float16* __restrict__ fn,
                                                    const int*      __restrict__ labels,
                                                    const float*    __restrict__ tempPtr,
                                                    float* __restrict__ rowSumExp,
                                                    float* __restrict__ rowSumPos) {
    __shared__ __align__(16) _Float16 As[TILE * DIM];   // 32 KB
    __shared__ __align__(16) _Float16 Bs[TILE * DIM];   // 32 KB  (64 KB total << 320 KB/WGP)

    const int j0  = blockIdx.x * TILE;
    const int i0  = blockIdx.y * TILE;
    const int tid = threadIdx.x;

    // Cooperative tile load: 2048 x 16B chunks per tile, 8 per thread, coalesced.
    #pragma unroll
    for (int k = 0; k < 8; ++k) {
        const int c   = tid + k * 256;
        const int row = c >> 4;            // 16 chunks per 128-half row
        const int col = (c & 15) * 8;
        *(v8h*)(As + row * DIM + col) = *(const v8h*)(fn + (i0 + row) * DIM + col);
        *(v8h*)(Bs + row * DIM + col) = *(const v8h*)(fn + (j0 + row) * DIM + col);
    }
    __syncthreads();

    const int w    = tid >> 5;      // wave 0..7 -> row strip
    const int lane = tid & 31;
    const int h    = lane >> 4;     // half-wave: selects K sub-range (A/B) and M range (C/D)
    const int ln   = lane & 15;     // M for A-frag; N for B/C/D-frags

    v8f acc[8] = {};                // 8 wave-tiles of 16x16 f32

    const int arow = w * 16 + ln;   // A-matrix: M = lane % 16 for both halves
    #pragma unroll
    for (int kk = 0; kk < 4; ++kk) {
        const int kb = kk * 32;
        // A 16-bit 16x32 layout: element e <-> K = (e&8)*2 + h*8 + (e&7)
        // -> elements 0..7  from [kb + h*8,      +8)
        //    elements 8..15 from [kb + 16 + h*8, +8)
        V16 a;
        a.h[0] = *(const v8h*)(As + arow * DIM + kb + h * 8);
        a.h[1] = *(const v8h*)(As + arow * DIM + kb + 16 + h * 8);
        #pragma unroll
        for (int t = 0; t < 8; ++t) {
            // B 32x16 layout: lane's column N = ln, K = (e&8)*2 + h*8 + (e&7)
            // (same as A) from row (j0 + t*16 + ln) of fn.
            V16 b;
            const _Float16* bp = Bs + (t * 16 + ln) * DIM + kb + h * 8;
            b.h[0] = *(const v8h*)(bp);
            b.h[1] = *(const v8h*)(bp + 16);
            acc[t] = __builtin_amdgcn_wmma_f32_16x16x32_f16(
                /*neg_a=*/false, a.v, /*neg_b=*/false, b.v,
                /*c_mod=*/(short)0, acc[t],
                /*reuse_a=*/false, /*reuse_b=*/false);
        }
    }

    // Epilogue. C/D layout: lane = h*16 + n holds N = n, VGPR r holds M = h*8 + r.
    const float invT = 1.0f / tempPtr[0];

    int   myLbl[8];
    #pragma unroll
    for (int r = 0; r < 8; ++r) myLbl[r] = labels[i0 + w * 16 + h * 8 + r];

    float eSum[8] = {0, 0, 0, 0, 0, 0, 0, 0};
    float pSum[8] = {0, 0, 0, 0, 0, 0, 0, 0};

    #pragma unroll
    for (int t = 0; t < 8; ++t) {
        const int jl = labels[j0 + t * 16 + ln];
        #pragma unroll
        for (int r = 0; r < 8; ++r) {
            const float logit = acc[t][r] * invT;
            const bool  same  = (myLbl[r] == jl);
            eSum[r] += same ? 0.0f : __expf(logit);   // denominator excludes same-label pairs
            pSum[r] += same ? logit : 0.0f;           // numerator: positive-pair logits
        }
    }

    // Reduce the 16 columns of each row across the lane half-group, then one
    // f32 atomic per row per (quantity, block).
    #pragma unroll
    for (int r = 0; r < 8; ++r) {
        float e = eSum[r], p = pSum[r];
        #pragma unroll
        for (int off = 8; off > 0; off >>= 1) {
            e += __shfl_xor(e, off, 32);   // xor of bits 0..3 stays within the half-group
            p += __shfl_xor(p, off, 32);
        }
        if (ln == 0) {
            const int i = i0 + w * 16 + h * 8 + r;
            atomicAdd(&rowSumExp[i], e);
            atomicAdd(&rowSumPos[i], p);
        }
    }
}

// ---------------------------------------------------------------------------
// Kernel 3: loss = -mean_i( sum_pos(i)/n_pos(i) - log(sum_exp(i)) ).
// Single block -> deterministic write of the scalar output (no atomics).
// ---------------------------------------------------------------------------
__global__ __launch_bounds__(256) void k_finalize(const float* __restrict__ rowSumExp,
                                                  const float* __restrict__ rowSumPos,
                                                  const int*   __restrict__ labels,
                                                  const int*   __restrict__ cnt,
                                                  float*       __restrict__ out) {
    __shared__ float sm[256];
    float p = 0.0f;
    for (int i = threadIdx.x; i < NROWS; i += 256) {
        const float np = (float)cnt[labels[i]];
        p += rowSumPos[i] / np - __logf(rowSumExp[i]);
    }
    sm[threadIdx.x] = p;
    __syncthreads();
    #pragma unroll
    for (int s = 128; s > 0; s >>= 1) {
        if (threadIdx.x < s) sm[threadIdx.x] += sm[threadIdx.x + s];
        __syncthreads();
    }
    if (threadIdx.x == 0) out[0] = -sm[0] / (float)NROWS;
}

// ---------------------------------------------------------------------------
extern "C" void kernel_launch(void* const* d_in, const int* in_sizes, int n_in,
                              void* d_out, int out_size, void* d_ws, size_t ws_size,
                              hipStream_t stream) {
    const float* feat    = (const float*)d_in[0];   // [8192,128] f32
    const int*   labels  = (const int*)d_in[1];     // [8192] int
    const float* tempPtr = (const float*)d_in[2];   // scalar
    float*       out     = (float*)d_out;

    // Workspace carve-up (~2.13 MB total):
    _Float16* fn        = (_Float16*)d_ws;                              // 8192*128 f16 = 2 MB
    float*    rowSumExp = (float*)((char*)d_ws + (size_t)NROWS * DIM * sizeof(_Float16));
    float*    rowSumPos = rowSumExp + NROWS;
    int*      cnt       = (int*)(rowSumPos + NROWS);                    // 256 bins

    k_zero<<<(NROWS + 255) / 256, 256, 0, stream>>>(rowSumExp, rowSumPos, cnt);
    k_prep<<<NROWS / 8, 256, 0, stream>>>(feat, labels, fn, cnt);
    k_gemm_fused<<<dim3(NT, NT), 256, 0, stream>>>(fn, labels, tempPtr, rowSumExp, rowSumPos);
    k_finalize<<<1, 256, 0, stream>>>(rowSumExp, rowSumPos, labels, cnt, out);
}
